// NeuralCompressor_81020263071867
// MI455X (gfx1250) — matrix operand order repaired and checked by reading, not compiled, not run on hardware
//
#include <hip/hip_runtime.h>
#include <hip/hip_bf16.h>

typedef __attribute__((ext_vector_type(16))) __bf16 v16bf;
typedef __attribute__((ext_vector_type(8)))  float  v8f;
typedef int v4i __attribute__((vector_size(16)));

#define RD 256
#define RCODES 1024
#define NQ 8
#define MROWS 16384

// ---------------------------------------------------------------- utilities
__device__ __forceinline__ float gelu_exact(float x) {
    return 0.5f * x * (1.0f + erff(x * 0.70710678118654752f));
}

__device__ __forceinline__ unsigned short f2bf(float f) {
    union { __bf16 h; unsigned short u; } c; c.h = (__bf16)f; return c.u;
}

union BU { v16bf v; unsigned int u[8]; };

// ------------------------------------------------- global->LDS async copies
#if defined(__has_builtin)
#if __has_builtin(__builtin_amdgcn_global_load_async_to_lds_b128)
#define HAVE_ASYNC_G2L 1
#endif
#endif
#ifndef HAVE_ASYNC_G2L
#define HAVE_ASYNC_G2L 0
#endif

__device__ __forceinline__ void copy16_g2l(void* lds, const void* g) {
#if HAVE_ASYNC_G2L
    __builtin_amdgcn_global_load_async_to_lds_b128(
        (__attribute__((address_space(1))) v4i*)(__UINTPTR_TYPE__)g,
        (__attribute__((address_space(3))) v4i*)(unsigned int)(__UINTPTR_TYPE__)lds,
        0, 0);
#else
    *(uint4*)lds = *(const uint4*)g;
#endif
}

template <int N>
__device__ __forceinline__ void wait_async_n() {
#if HAVE_ASYNC_G2L
#if __has_builtin(__builtin_amdgcn_s_wait_asynccnt)
    __builtin_amdgcn_s_wait_asynccnt(N);
#else
    asm volatile("s_wait_asynccnt %0" :: "n"(N) : "memory");
#endif
#endif
}

// ------------------------------------------------- WMMA fragment loads (LDS)
// Tiles stored [row][K] bf16; uints are K-pairs, bit-cast straight into the
// operand order k = 16*half + s (same for A and B): pure ds_load_b128, no unpack VALU.
__device__ __forceinline__ v16bf load_a_frag(const unsigned int* tile, int ldu, int lane) {
    const int row  = lane & 15;
    const int half = lane >> 4;
    const unsigned int* p = tile + row * ldu + half * 8;
    BU r;
#pragma unroll
    for (int j = 0; j < 8; ++j) r.u[j] = p[j];
    return r.v;
}

__device__ __forceinline__ v16bf load_b_frag(const unsigned int* tile, int ldu, int lane) {
    const int col  = lane & 15;
    const int half = lane >> 4;
    const unsigned int* p = tile + col * ldu + half * 8;
    BU r;
#pragma unroll
    for (int j = 0; j < 8; ++j) r.u[j] = p[j];
    return r.v;
}

__device__ __forceinline__ v8f wmma_bf16(v16bf a, v16bf b, v8f c) {
    return __builtin_amdgcn_wmma_f32_16x16x32_bf16(false, a, false, b, (short)0, c, false, false);
}

// ---------------- NT GEMM: C[M,N] = act(A[M,K] @ Bt[N,K]^T + bias) ---------
// Block tile 128x64, 8 waves of 32x32 (4 WMMA accumulators each),
// double-buffered LDS with async global->LDS staging.
#define GTM 128
#define GTN 64
#define TK 32
#define LDT 40          // ushort stride (80B, 16B-multiple)
#define LDU (LDT / 2)   // uint stride

__global__ __launch_bounds__(256) void gemm_nt_bf16_kernel(
    const unsigned short* __restrict__ A,   // [M][K] bf16
    const unsigned short* __restrict__ Bt,  // [N][K] bf16
    const float* __restrict__ bias,
    float* __restrict__ Cf, unsigned short* __restrict__ Cbf,
    int M, int N, int K, int act)
{
    __shared__ unsigned short As[2][GTM * LDT];
    __shared__ unsigned short Bs[2][GTN * LDT];

    const int tid  = threadIdx.x;
    const int lane = tid & 31;
    const int wave = tid >> 5;
    const int wr   = wave >> 1;   // 0..3 : 32-row band
    const int wc   = wave & 1;    // 0..1 : 32-col band
    const int m0   = blockIdx.y * GTM;
    const int n0   = blockIdx.x * GTN;

    const int sr = tid >> 2;          // 0..63
    const int sc = (tid & 3) * 8;     // 0..24 (bf16)

    v8f acc0 = {}, acc1 = {}, acc2 = {}, acc3 = {};

    const int nk = K / TK;

    // per-tile staging: 3 async b128 copies per thread (A: 2 rows, B: 1 row)
    auto stage = [&](int buf, int k0) {
        copy16_g2l(&As[buf][sr * LDT + sc],        &A [(size_t)(m0 + sr)      * K + k0 + sc]);
        copy16_g2l(&As[buf][(sr + 64) * LDT + sc], &A [(size_t)(m0 + sr + 64) * K + k0 + sc]);
        copy16_g2l(&Bs[buf][sr * LDT + sc],        &Bt[(size_t)(n0 + sr)      * K + k0 + sc]);
    };

    stage(0, 0);
    for (int t = 0; t < nk; ++t) {
        if (t + 1 < nk) { stage((t + 1) & 1, (t + 1) * TK); wait_async_n<3>(); }
        else            { wait_async_n<0>(); }
        if (t + 2 < nk) {
            __builtin_prefetch(&A [(size_t)(m0 + sr) * K + (t + 2) * TK + sc], 0, 1);
            __builtin_prefetch(&Bt[(size_t)(n0 + sr) * K + (t + 2) * TK + sc], 0, 1);
        }
        __syncthreads();

        const unsigned int* ab = (const unsigned int*)As[t & 1];
        const unsigned int* bb = (const unsigned int*)Bs[t & 1];
        v16bf a0 = load_a_frag(ab + (wr * 32)      * LDU, LDU, lane);
        v16bf a1 = load_a_frag(ab + (wr * 32 + 16) * LDU, LDU, lane);
        v16bf b0 = load_b_frag(bb + (wc * 32)      * LDU, LDU, lane);
        v16bf b1 = load_b_frag(bb + (wc * 32 + 16) * LDU, LDU, lane);
        acc0 = wmma_bf16(a0, b0, acc0);
        acc1 = wmma_bf16(a0, b1, acc1);
        acc2 = wmma_bf16(a1, b0, acc2);
        acc3 = wmma_bf16(a1, b1, acc3);
        __syncthreads();
    }

    const int cn = lane & 15;
    const int mh = (lane >> 4) * 8;
    v8f accs[4] = {acc0, acc1, acc2, acc3};
#pragma unroll
    for (int i = 0; i < 2; ++i) {
#pragma unroll
        for (int j = 0; j < 2; ++j) {
            v8f c = accs[i * 2 + j];
#pragma unroll
            for (int r = 0; r < 8; ++r) {
                int gm = m0 + wr * 32 + i * 16 + mh + r;
                int gn = n0 + wc * 32 + j * 16 + cn;
                float v = c[r] + bias[gn];
                if (act) v = gelu_exact(v);
                if (Cf) Cf[(size_t)gm * N + gn] = v;
                else    Cbf[(size_t)gm * N + gn] = f2bf(v);
            }
        }
    }
}

// ---------------- fused RVQ stage ------------------------------------------
#define LDA 264   // ushort stride of the 64x256 residual band (528B, 16B-mult)

__global__ __launch_bounds__(256) void rvq_stage_kernel(
    float* __restrict__ res, unsigned short* __restrict__ res_bf,
    float* __restrict__ qtot,
    const float* __restrict__ cb,             // fp32 [1024][256] (exact update)
    const unsigned short* __restrict__ cb_bf, // bf16 [1024][256]
    const float* __restrict__ cbn,            // [1024]
    int* __restrict__ idx_out,                // [16384]
    float* __restrict__ closs_part)           // [256]
{
    __shared__ unsigned short As[64 * LDA];
    __shared__ unsigned short Bs[2][64 * LDT];
    __shared__ float  sminv[2][64];
    __shared__ int    smini[2][64];
    __shared__ int    fidx[64];
    __shared__ float  rsum[256];

    const int tid  = threadIdx.x;
    const int lane = tid & 31;
    const int wave = tid >> 5;
    const int wr   = wave >> 1, wc = wave & 1;
    const int m0   = blockIdx.x * 64;

    // stage full residual band once: 64x256 bf16 (8 async b128 per thread)
#pragma unroll
    for (int e = 0; e < 8; ++e) {
        int id = e * 256 + tid;
        int r = id >> 5, c8 = (id & 31) * 8;
        copy16_g2l(&As[r * LDA + c8], &res_bf[(size_t)(m0 + r) * RD + c8]);
    }
    wait_async_n<0>();
    __syncthreads();

    float minv[8]; int mini[8];
#pragma unroll
    for (int r = 0; r < 8; ++r) { minv[r] = 3.4e38f; mini[r] = 0; }

    const int sr = tid >> 2;
    const int sc = (tid & 3) * 8;
    const unsigned int* Au = (const unsigned int*)As;

    v8f c0 = {}, c1 = {};

    // 128 linearized steps: step s -> code tile ct = s>>3, k0 = (s&7)*32
    copy16_g2l(&Bs[0][sr * LDT + sc], &cb_bf[(size_t)sr * RD + sc]);
    for (int s = 0; s < 128; ++s) {
        const int ct = s >> 3, k0 = (s & 7) * TK;
        if (s + 1 < 128) {
            const int ct1 = (s + 1) >> 3, k01 = ((s + 1) & 7) * TK;
            copy16_g2l(&Bs[(s + 1) & 1][sr * LDT + sc],
                       &cb_bf[(size_t)(ct1 * 64 + sr) * RD + k01 + sc]);
            wait_async_n<1>();
        } else {
            wait_async_n<0>();
        }
        __syncthreads();

        if ((s & 7) == 0) { v8f z = {}; c0 = z; c1 = z; }
        const unsigned int* bb = (const unsigned int*)Bs[s & 1];
        v16bf af = load_a_frag(Au + (wr * 16) * (LDA / 2) + (k0 >> 1), LDA / 2, lane);
        v16bf b0 = load_b_frag(bb + (wc * 32)      * LDU, LDU, lane);
        v16bf b1 = load_b_frag(bb + (wc * 32 + 16) * LDU, LDU, lane);
        c0 = wmma_bf16(af, b0, c0);
        c1 = wmma_bf16(af, b1, c1);
        __syncthreads();

        if ((s & 7) == 7) {
            const int code0 = ct * 64;
            const int cc = lane & 15;
#pragma unroll
            for (int r = 0; r < 8; ++r) {
                int ca = code0 + wc * 32 + cc;
                float da = cbn[ca] - 2.0f * c0[r];
                if (da < minv[r] || (da == minv[r] && ca < mini[r])) { minv[r] = da; mini[r] = ca; }
                int cb2 = ca + 16;
                float db = cbn[cb2] - 2.0f * c1[r];
                if (db < minv[r] || (db == minv[r] && cb2 < mini[r])) { minv[r] = db; mini[r] = cb2; }
            }
        }
    }

    // per-row argmin: xor masks 1..8 stay inside each 16-lane half (same rows)
#pragma unroll
    for (int r = 0; r < 8; ++r) {
#pragma unroll
        for (int mask = 8; mask >= 1; mask >>= 1) {
            float ov = __shfl_xor(minv[r], mask, 32);
            int   oi = __shfl_xor(mini[r], mask, 32);
            if (ov < minv[r] || (ov == minv[r] && oi < mini[r])) { minv[r] = ov; mini[r] = oi; }
        }
        if ((lane & 15) == 0) {
            int row = wr * 16 + (lane >> 4) * 8 + r;
            sminv[wc][row] = minv[r];
            smini[wc][row] = mini[r];
        }
    }
    __syncthreads();
    if (tid < 64) {
        float v0 = sminv[0][tid]; int i0 = smini[0][tid];
        float v1 = sminv[1][tid]; int i1 = smini[1][tid];
        int best = (v1 < v0 || (v1 == v0 && i1 < i0)) ? i1 : i0;
        fidx[tid] = best;
        idx_out[m0 + tid] = best;
    }
    __syncthreads();

    // exact fp32 residual/q_total update + bf16 mirror + commitment partial
    float lsq = 0.0f;
    const int row  = tid >> 2;
    const int d0   = (tid & 3) * 64;
    const int code = fidx[row];
    for (int d = 0; d < 64; ++d) {
        size_t g = (size_t)(m0 + row) * RD + d0 + d;
        float q  = cb[(size_t)code * RD + d0 + d];
        float rv = res[g] - q;
        res[g]    = rv;
        res_bf[g] = f2bf(rv);
        qtot[g]  += q;
        lsq += rv * rv;
    }
    rsum[tid] = lsq;
    __syncthreads();
    for (int t = 128; t > 0; t >>= 1) {
        if (tid < t) rsum[tid] += rsum[tid + t];
        __syncthreads();
    }
    if (tid == 0) closs_part[blockIdx.x] = rsum[0];
}

// ---------------- small helpers --------------------------------------------
__global__ void zero_kernel(float* p, int n) {
    int i = blockIdx.x * 256 + threadIdx.x;
    if (i < n) p[i] = 0.0f;
}

__global__ void cvt_bf16_kernel(const float* __restrict__ src, unsigned short* __restrict__ dst, int n) {
    int i = blockIdx.x * 256 + threadIdx.x;
    if (i < n) dst[i] = f2bf(src[i]);
}

// src [K][N] fp32 -> dst [N][K] bf16 (one-time weight transpose)
__global__ void transpose_cvt_kernel(const float* __restrict__ src, unsigned short* __restrict__ dst, int K, int N) {
    int i = blockIdx.x * 256 + threadIdx.x;
    if (i < K * N) {
        int k = i / N, n = i % N;
        dst[(size_t)n * K + k] = f2bf(src[i]);
    }
}

__global__ void cbnorm_kernel(const float* __restrict__ cb, float* __restrict__ cbn, int total) {
    int i = blockIdx.x * 256 + threadIdx.x;
    if (i < total) {
        const float* p = cb + (size_t)i * RD;
        float s = 0.0f;
        for (int d = 0; d < RD; ++d) s += p[d] * p[d];
        cbn[i] = s;
    }
}

__global__ void commit_kernel(const float* __restrict__ part, float* __restrict__ out) {
    __shared__ float s[256];
    float v = 0.0f;
    for (int i = threadIdx.x; i < NQ * 256; i += 256) v += part[i];
    s[threadIdx.x] = v;
    __syncthreads();
    for (int t = 128; t > 0; t >>= 1) {
        if (threadIdx.x < t) s[threadIdx.x] += s[threadIdx.x + t];
        __syncthreads();
    }
    if (threadIdx.x == 0)
        out[0] = 0.25f * s[0] / (8.0f * 4194304.0f);
}

// ---------------- launch ----------------------------------------------------
extern "C" void kernel_launch(void* const* d_in, const int* in_sizes, int n_in,
                              void* d_out, int out_size, void* d_ws, size_t ws_size,
                              hipStream_t stream) {
    const float* x   = (const float*)d_in[0];
    const float* We1 = (const float*)d_in[1];
    const float* be1 = (const float*)d_in[2];
    const float* We2 = (const float*)d_in[3];
    const float* be2 = (const float*)d_in[4];
    const float* cbs = (const float*)d_in[5];
    const float* Wd1 = (const float*)d_in[6];
    const float* bd1 = (const float*)d_in[7];
    const float* Wd2 = (const float*)d_in[8];
    const float* bd2 = (const float*)d_in[9];

    float* out        = (float*)d_out;
    float* z_out      = out;
    float* q_out      = out + 4194304;
    int*   idx_out    = (int*)(out + 8388608);
    float* rec_out    = out + 8519680;
    float* commit_out = out + 25296896;

    float* ws = (float*)d_ws;
    unsigned short* bufbf   = (unsigned short*)(ws);              // 16.7M bf16
    float* res   = ws + 8388608;
    float* qtot  = ws + 12582912;
    float* cbn   = ws + 16777216;
    float* clp   = ws + 16785408;
    unsigned short* x_bf    = (unsigned short*)(ws + 16787456);   // 16.7M bf16
    unsigned short* We1t    = (unsigned short*)(ws + 25176064);   // [1024][1024]
    unsigned short* We2t    = (unsigned short*)(ws + 25700352);   // [256][1024]
    unsigned short* Wd1t    = (unsigned short*)(ws + 25831424);   // [1024][256]
    unsigned short* Wd2t    = (unsigned short*)(ws + 25962496);   // [1024][1024]
    unsigned short* cb_bf   = (unsigned short*)(ws + 26486784);   // 8*1024*256
    unsigned short* res_bf  = (unsigned short*)(ws + 27535360);
    unsigned short* qtot_bf = (unsigned short*)(ws + 29632512);

    const int M = MROWS;

    zero_kernel<<<dim3(16425), 256, 0, stream>>>(qtot, 4204544);      // qtot+cbn+clp span
    cvt_bf16_kernel<<<dim3(65536), 256, 0, stream>>>(x, x_bf, 16777216);
    transpose_cvt_kernel<<<dim3(4096), 256, 0, stream>>>(We1, We1t, 1024, 1024);
    transpose_cvt_kernel<<<dim3(1024), 256, 0, stream>>>(We2, We2t, 1024, 256);
    transpose_cvt_kernel<<<dim3(1024), 256, 0, stream>>>(Wd1, Wd1t, 256, 1024);
    transpose_cvt_kernel<<<dim3(4096), 256, 0, stream>>>(Wd2, Wd2t, 1024, 1024);
    cvt_bf16_kernel<<<dim3(8192), 256, 0, stream>>>(cbs, cb_bf, 2097152);
    cbnorm_kernel<<<dim3(32), 256, 0, stream>>>(cbs, cbn, NQ * RCODES);

    // encoder
    gemm_nt_bf16_kernel<<<dim3(1024 / GTN, M / GTM), 256, 0, stream>>>(
        x_bf, We1t, be1, nullptr, bufbf, M, 1024, 1024, 1);
    gemm_nt_bf16_kernel<<<dim3(RD / GTN, M / GTM), 256, 0, stream>>>(
        bufbf, We2t, be2, z_out, nullptr, M, RD, 1024, 0);

    (void)hipMemcpyAsync(res, z_out, (size_t)4194304 * sizeof(float), hipMemcpyDeviceToDevice, stream);
    cvt_bf16_kernel<<<dim3(16384), 256, 0, stream>>>(z_out, res_bf, 4194304);

    for (int s = 0; s < NQ; ++s) {
        rvq_stage_kernel<<<dim3(M / 64), 256, 0, stream>>>(
            res, res_bf, qtot,
            cbs + (size_t)s * RCODES * RD, cb_bf + (size_t)s * RCODES * RD,
            cbn + s * RCODES, idx_out + s * M, clp + s * 256);
    }

    (void)hipMemcpyAsync(q_out, qtot, (size_t)4194304 * sizeof(float), hipMemcpyDeviceToDevice, stream);
    cvt_bf16_kernel<<<dim3(16384), 256, 0, stream>>>(qtot, qtot_bf, 4194304);

    // decoder
    gemm_nt_bf16_kernel<<<dim3(1024 / GTN, M / GTM), 256, 0, stream>>>(
        qtot_bf, Wd1t, bd1, nullptr, bufbf, M, 1024, RD, 1);
    gemm_nt_bf16_kernel<<<dim3(1024 / GTN, M / GTM), 256, 0, stream>>>(
        bufbf, Wd2t, bd2, rec_out, nullptr, M, 1024, 1024, 0);

    commit_kernel<<<dim3(1), 256, 0, stream>>>(clp, commit_out);
}
